// GCN_11098195493584
// MI455X (gfx1250) — matrix operand order repaired and checked by reading, not compiled, not run on hardware
//
#include <hip/hip_runtime.h>
#include <hip/hip_bf16.h>

// ---------------------------------------------------------------------------
// GCN (4x GCNConv + BN + ReLU + residual) -> full-sequence MHA -> out proj
// CDNA5 / gfx1250: wave32, v_wmma_f32_16x16x32_f16 for all dense matmuls.
// Attention: transposed scores (K x Q^T) so exp'd scores are already the WMMA
// A-fragment for P x V; f16 qkv buffer; double-buffered LDS K/V tiles with
// async global->LDS staging (ASYNCcnt).
// ---------------------------------------------------------------------------

#define NN   8192
#define EE   262144
#define DIN  64
#define HID  128
#define DOUT 16
#define NHEAD 4
#define HD   32            // head dim
#define EPS  1e-5f
#define KVS  40            // padded LDS row stride (halves) for 32-wide tiles

#if defined(__has_builtin)
#if __has_builtin(__builtin_amdgcn_global_load_async_to_lds_b128)
#define HAVE_ASYNC_LDS 1
#endif
#endif

#if defined(__has_builtin)
#if __has_builtin(__builtin_amdgcn_exp2f)
#define FAST_EXP2(x) __builtin_amdgcn_exp2f(x)   // raw v_exp_f32 (2^x), FTZ
#else
#define FAST_EXP2(x) __expf((x) * 0.6931471805599453f)
#endif
#else
#define FAST_EXP2(x) __expf((x) * 0.6931471805599453f)
#endif

typedef __attribute__((ext_vector_type(16))) _Float16 v16h;
typedef __attribute__((ext_vector_type(8)))  _Float16 v8h;
typedef __attribute__((ext_vector_type(8)))  float    v8f;
typedef __attribute__((ext_vector_type(4)))  int      v4i;

// ---------------- WMMA fragment helpers (CDNA5 16x16x32 f16 layouts) -------
// A (16x32 f16): lane L holds row M=L&15.
//   halves 0..7  : K = ((L&16)?8:0)  + i
//   halves 8..15 : K = 16 + ((L&16)?8:0) + i
__device__ __forceinline__ v16h frag_a_f32(const float* __restrict__ A, int lda, int lane) {
    const int m  = lane & 15;
    const int b0 = (lane & 16) ? 8 : 0;
    const float* p = A + m * lda;
    v16h a;
#pragma unroll
    for (int i = 0; i < 8; ++i) a[i]     = (_Float16)p[b0 + i];
#pragma unroll
    for (int i = 0; i < 8; ++i) a[8 + i] = (_Float16)p[16 + b0 + i];
    return a;
}

// A fragment from an LDS row-major f16 tile (stride in halves)
__device__ __forceinline__ v16h frag_a_lds(const _Float16* __restrict__ src,
                                           int stride, int lane) {
    const int m  = lane & 15;
    const int b0 = (lane & 16) ? 8 : 0;
    const _Float16* p = src + m * stride + b0;
    v16h a;
#pragma unroll
    for (int i = 0; i < 8; ++i) a[i]     = p[i];
#pragma unroll
    for (int i = 0; i < 8; ++i) a[8 + i] = p[16 + i];
    return a;
}

// B fragment (32 x 16) from LDS, column-major panel: src[n*stride + k]
__device__ __forceinline__ v16h frag_b_lds(const _Float16* __restrict__ src,
                                           int stride, int lane) {
    const int n  = lane & 15;
    const int kb = lane & 16;
    const _Float16* p = src + n * stride + (kb >> 1);
    v16h b;
#pragma unroll
    for (int i = 0; i < 16; ++i) b[i] = p[i + (i & 8)];
    return b;
}

// ---------------- generic WMMA GEMM: C[MxNc] = A[MxK] * B(+bias) -----------
// grid: (Nc/16, M/64), block 128 threads (4 waves, one 16x16 tile each).
// The 16-column B panel is shared by all 4 waves -> staged once in LDS (f16).
// OutT = float or _Float16 (f16 path feeds attention directly).
template <int TRANSB, typename OutT>
__global__ void __launch_bounds__(128)
gemm_wmma_kernel(const float* __restrict__ A, const float* __restrict__ B,
                 const float* __restrict__ bias, OutT* __restrict__ C,
                 int M, int Nc, int K)
{
    __shared__ _Float16 LB[16 * (HID + 8)];        // max K=128, stride K+8
    const int lane = threadIdx.x & 31;
    const int wave = threadIdx.x >> 5;
    const int row0 = blockIdx.y * 64 + wave * 16;
    const int col0 = blockIdx.x * 16;
    const int KS   = K + 8;

    // ---- stage B panel (16 cols x K) into LDS, col-major f16
    {
        const int col   = threadIdx.x & 15;
        const int chunk = threadIdx.x >> 4;        // 0..7
        const int kpt   = K >> 3;                  // elems per thread
        for (int j = 0; j < kpt; ++j) {
            int k = chunk * kpt + j;
            float v = TRANSB ? B[(col0 + col) * K + k]
                             : B[k * Nc + col0 + col];
            LB[col * KS + k] = (_Float16)v;
        }
    }
    __syncthreads();

    v8f acc = {};
    for (int k0 = 0; k0 < K; k0 += 32) {
        v16h a = frag_a_f32(A + row0 * K + k0, K, lane);
        v16h b = frag_b_lds(LB + k0, KS, lane);
        acc = __builtin_amdgcn_wmma_f32_16x16x32_f16(
                  false, a, false, b, (short)0, acc, false, false);
    }
    const int n  = lane & 15;
    const int mb = row0 + ((lane & 16) ? 8 : 0);
    const float bv = bias ? bias[col0 + n] : 0.0f;
#pragma unroll
    for (int r = 0; r < 8; ++r)
        C[(mb + r) * Nc + col0 + n] = (OutT)(acc[r] + bv);
}

// ---------------- degree / normalization -----------------------------------
__global__ void deg_init_kernel(float* d) {
    int i = blockIdx.x * 256 + threadIdx.x;
    if (i < NN) d[i] = 1.0f;                       // self loop
}
__global__ void deg_edge_kernel(const long long* __restrict__ ei, float* d) {
    int e = blockIdx.x * 256 + threadIdx.x;
    if (e < EE) atomicAdd(&d[(int)ei[EE + e]], 1.0f);
}
__global__ void deg_rsqrt_kernel(float* d) {
    int i = blockIdx.x * 256 + threadIdx.x;
    if (i < NN) d[i] = rsqrtf(d[i]);
}

// ---------------- aggregation ----------------------------------------------
__global__ void bias_init_kernel(const float* __restrict__ b, float* __restrict__ agg) {
    int idx = blockIdx.x * 256 + threadIdx.x;
    agg[idx] = b[idx & (HID - 1)];
}
// one wave per edge, 4 columns per lane (coalesced 512B per message row);
// atomics land in L2 (h@W is 4MB << 192MB L2)
__global__ void __launch_bounds__(256)
scatter_kernel(const long long* __restrict__ ei, const float* __restrict__ t,
               const float* __restrict__ dinv, float* __restrict__ agg)
{
    const int lane = threadIdx.x & 31;
    const int e    = blockIdx.x * 8 + (threadIdx.x >> 5);
    const int s = (int)ei[e];
    const int d = (int)ei[EE + e];
    const float w = dinv[s] * dinv[d];
    const float* src = t + s * HID;
    float* dst = agg + d * HID;
#pragma unroll
    for (int j = 0; j < 4; ++j) {
        int c = lane + j * 32;
        atomicAdd(&dst[c], src[c] * w);
    }
}
__global__ void selfloop_kernel(const float* __restrict__ t,
                                const float* __restrict__ dinv,
                                float* __restrict__ agg)
{
    int idx = blockIdx.x * 256 + threadIdx.x;
    float di = dinv[idx >> 7];
    agg[idx] += t[idx] * di * di;
}

// ---------------- batchnorm ------------------------------------------------
__global__ void stats_zero_kernel(float* s) { s[threadIdx.x] = 0.0f; }      // 256 thr
__global__ void __launch_bounds__(128)
stats_partial_kernel(const float* __restrict__ h, float* __restrict__ s)
{
    const int c = threadIdx.x;
    float sum = 0.0f, sq = 0.0f;
    for (int r = blockIdx.x; r < NN; r += gridDim.x) {
        float v = h[r * HID + c];
        sum += v; sq += v * v;
    }
    atomicAdd(&s[c], sum);
    atomicAdd(&s[HID + c], sq);
}
__global__ void stats_final_kernel(float* s) {                              // 128 thr
    int c = threadIdx.x;
    float mu  = s[c] * (1.0f / NN);
    float var = s[HID + c] * (1.0f / NN) - mu * mu;
    s[2 * HID + c] = mu;
    s[3 * HID + c] = rsqrtf(var + EPS);
}
__global__ void bn_apply_kernel(const float* __restrict__ h, const float* __restrict__ s,
                                const float* __restrict__ g, const float* __restrict__ be,
                                const float* __restrict__ add,
                                float* __restrict__ out, float* __restrict__ out2)
{
    int idx = blockIdx.x * 256 + threadIdx.x;
    int c = idx & (HID - 1);
    float v = g[c] * (h[idx] - s[2 * HID + c]) * s[3 * HID + c] + be[c];
    if (add) v += add[idx];
    v = fmaxf(v, 0.0f);
    out[idx] = v;
    if (out2) out2[idx] = v;
}

// ---------------- flash attention (transposed scores, f16 qkv) -------------
// grid (NN/128, NHEAD), 128 threads = 4 waves; wave = two 16-row Q tiles.
// Scores computed as S' = K_tile x Q^T so exp(S') is directly the WMMA
// A-fragment for P x V. Fixed-shift softmax (scores bounded << 88): no
// running max, no rescale, no cross-lane reductions in the loop.
// K/V tiles double-buffered in LDS; K tile filled by async global->LDS DMA
// (ASYNCcnt), retired just before the per-iteration barrier.
__global__ void __launch_bounds__(128)
attn_kernel(const _Float16* __restrict__ qh, float* __restrict__ ctx)
{
    __shared__ _Float16 Ksh[2][32 * KVS];    // [key row][d within head]
    __shared__ _Float16 Vsh[2][32 * KVS];    // [head col][key row] (transposed)
    __shared__ float    Lsh[4][32];          // per-wave 1/l broadcast (end only)
    const int lane = threadIdx.x & 31;
    const int wave = threadIdx.x >> 5;
    const int head = blockIdx.y;
    const int q0   = (blockIdx.x * 4 + wave) * 32;   // two 16-row tiles
    const int hoff = head * HD;
    // 1/sqrt(32) * log2(e): single v_mul feeding v_exp_f32 (= 2^x)
    const float sc2 = 0.17677669529663687f * 1.4426950408889634f;

    const int n  = lane & 15;
    const int kb = lane & 16;

    // Q as B-fragments (32d x 16 qrows), straight f16 copy, live in regs
    v16h bq[2];
#pragma unroll
    for (int t = 0; t < 2; ++t) {
        const _Float16* qp = qh + (q0 + t * 16 + n) * 384 + hoff + (kb >> 1);
#pragma unroll
        for (int i = 0; i < 16; ++i) bq[t][i] = qp[i + (i & 8)];
    }

    v8f acc0[2] = {{}, {}}, acc1[2] = {{}, {}};
    float lsum[2] = {0.0f, 0.0f};

    // staging assignment: thread -> (key row, 8-half chunk)
    const int sr = threadIdx.x >> 2;          // 0..31
    const int sc = (threadIdx.x & 3) * 8;     // 0,8,16,24

    auto stage = [&](int kv, int buf) {
        const _Float16* kp = qh + (kv + sr) * 384 + HID + hoff + sc;
        const _Float16* vp = qh + (kv + sr) * 384 + 2 * HID + hoff + sc;
#ifdef HAVE_ASYNC_LDS
        // 16B/lane async DMA into LDS, tracked on ASYNCcnt
        __builtin_amdgcn_global_load_async_to_lds_b128(
            (__attribute__((address_space(1))) v4i*)kp,
            (__attribute__((address_space(3))) v4i*)&Ksh[buf][sr * KVS + sc],
            0, 0);
#else
        v8h kk = *(const v8h*)kp;
        *(v8h*)&Ksh[buf][sr * KVS + sc] = kk;           // row-major copy
#endif
        v8h vv = *(const v8h*)vp;
#pragma unroll
        for (int j = 0; j < 8; ++j)                      // transpose V
            Vsh[buf][(sc + j) * KVS + sr] = vv[j];
        if (kv + 32 < NN) {                              // hint tile after next
            __builtin_prefetch(vp + 32 * 384, 0, 3);
        }
    };

    auto stage_retire = [&]() {
#ifdef HAVE_ASYNC_LDS
#if __has_builtin(__builtin_amdgcn_s_wait_asynccnt)
        __builtin_amdgcn_s_wait_asynccnt(0);
#else
        asm volatile("s_wait_asynccnt 0x0" ::: "memory");
#endif
#endif
    };

    stage(0, 0);
    stage_retire();
    __syncthreads();

    for (int kv = 0; kv < NN; kv += 32) {
        const int buf = (kv >> 5) & 1;
        if (kv + 32 < NN) stage(kv + 32, buf ^ 1);   // overlap next tile's fill

        // K tiles as A-fragments (keys on the M axis)
        v16h ak0 = frag_a_lds(Ksh[buf], KVS, lane);            // keys kv..kv+15
        v16h ak1 = frag_a_lds(Ksh[buf] + 16 * KVS, KVS, lane); // keys kv+16..+31
        // V tiles as B-fragments (from transposed LDS tile)
        v16h bv0 = frag_b_lds(Vsh[buf], KVS, lane);            // d cols 0..15
        v16h bv1 = frag_b_lds(Vsh[buf] + 16 * KVS, KVS, lane); // d cols 16..31

#pragma unroll
        for (int t = 0; t < 2; ++t) {
            v8f z = {};
            // S'[key][qrow]: lane = qrow, VGPR r = key (r / 8+r by lane group)
            v8f sa = __builtin_amdgcn_wmma_f32_16x16x32_f16(false, ak0, false, bq[t], (short)0, z, false, false);
            v8f sb = __builtin_amdgcn_wmma_f32_16x16x32_f16(false, ak1, false, bq[t], (short)0, z, false, false);

            // 2^(s * scale*log2e) == exp(s*scale); raw v_exp_f32, one mul each.
            // Result IS the A-fragment of P for this q tile.
            v16h ap;
            float ls = 0.0f;
#pragma unroll
            for (int i = 0; i < 8; ++i) {
                float ea = FAST_EXP2(sa[i] * sc2);
                float eb = FAST_EXP2(sb[i] * sc2);
                ap[i]     = (_Float16)ea;
                ap[8 + i] = (_Float16)eb;
                ls += ea + eb;
            }
            lsum[t] += ls;

            acc0[t] = __builtin_amdgcn_wmma_f32_16x16x32_f16(false, ap, false, bv0, (short)0, acc0[t], false, false);
            acc1[t] = __builtin_amdgcn_wmma_f32_16x16x32_f16(false, ap, false, bv1, (short)0, acc1[t], false, false);
        }
        stage_retire();    // async fill of buf^1 complete (wave-local)
        __syncthreads();   // block-wide: buf^1 valid, buf free for next staging
    }

    // combine the two key-halves of each row sum (lane L <-> L^16), then
    // broadcast 1/l from lane-indexed to VGPR-indexed layout via tiny LDS
#pragma unroll
    for (int t = 0; t < 2; ++t)
        lsum[t] += __shfl_xor(lsum[t], 16, 32);
    if (lane < 16) {
        Lsh[wave][lane]      = 1.0f / lsum[0];
        Lsh[wave][16 + lane] = 1.0f / lsum[1];
    }
    asm volatile("s_wait_dscnt 0" ::: "memory");   // intra-wave LDS RAW

    const int mb = (lane & 16) ? 8 : 0;
#pragma unroll
    for (int t = 0; t < 2; ++t) {
#pragma unroll
        for (int r = 0; r < 8; ++r) {
            float inv = Lsh[wave][t * 16 + mb + r];
            int row = q0 + t * 16 + mb + r;
            ctx[row * HID + hoff + n]      = acc0[t][r] * inv;
            ctx[row * HID + hoff + 16 + n] = acc1[t][r] * inv;
        }
    }
}

// ---------------------------------------------------------------------------
extern "C" void kernel_launch(void* const* d_in, const int* in_sizes, int n_in,
                              void* d_out, int out_size, void* d_ws, size_t ws_size,
                              hipStream_t stream)
{
    const float*     X    = (const float*)d_in[0];
    const long long* EI   = (const long long*)d_in[1];
    const float* W[4]  = {(const float*)d_in[2],  (const float*)d_in[6],
                          (const float*)d_in[10], (const float*)d_in[14]};
    const float* B[4]  = {(const float*)d_in[3],  (const float*)d_in[7],
                          (const float*)d_in[11], (const float*)d_in[15]};
    const float* G[4]  = {(const float*)d_in[4],  (const float*)d_in[8],
                          (const float*)d_in[12], (const float*)d_in[16]};
    const float* BE[4] = {(const float*)d_in[5],  (const float*)d_in[9],
                          (const float*)d_in[13], (const float*)d_in[17]};
    const float* resW  = (const float*)d_in[18];
    const float* resB  = (const float*)d_in[19];
    const float* inW   = (const float*)d_in[20];   // [384,128]
    const float* inB   = (const float*)d_in[21];
    const float* opW   = (const float*)d_in[22];   // [128,128]
    const float* opB   = (const float*)d_in[23];
    const float* outW  = (const float*)d_in[24];   // [128,16]
    const float* outB  = (const float*)d_in[25];
    float* OUT = (float*)d_out;

    // workspace carving
    float* ws    = (float*)d_ws;
    float* dinv  = ws;                          // NN
    float* stats = dinv + NN;                   // 512
    float* t     = stats + 512;                 // NN*HID (conv linear out; later ctx)
    float* agg   = t + (size_t)NN * HID;        // NN*HID
    float* h     = agg + (size_t)NN * HID;      // NN*HID
    float* resid = h + (size_t)NN * HID;        // NN*HID
    float* rlin  = resid + (size_t)NN * HID;    // NN*HID
    _Float16* qh = (_Float16*)(rlin + (size_t)NN * HID);   // NN*384 halves
    float* ctx   = t;                           // alias: t is dead after layer 4

    const dim3 blk128(128), blk256(256);

    // degree normalization: deg starts at 1 (self loops), dinv = rsqrt(deg)
    deg_init_kernel <<<NN / 256, blk256, 0, stream>>>(dinv);
    deg_edge_kernel <<<EE / 256, blk256, 0, stream>>>(EI, dinv);
    deg_rsqrt_kernel<<<NN / 256, blk256, 0, stream>>>(dinv);

    auto conv_bn = [&](const float* hin, int K, int layer, const float* add,
                       float* hout, float* hdup) {
        gemm_wmma_kernel<0, float><<<dim3(HID / 16, NN / 64), blk128, 0, stream>>>(
            hin, W[layer], nullptr, t, NN, HID, K);
        bias_init_kernel<<<NN * HID / 256, blk256, 0, stream>>>(B[layer], agg);
        scatter_kernel  <<<EE / 8, blk256, 0, stream>>>(EI, t, dinv, agg);
        selfloop_kernel <<<NN * HID / 256, blk256, 0, stream>>>(t, dinv, agg);
        stats_zero_kernel   <<<1, blk256, 0, stream>>>(stats);
        stats_partial_kernel<<<64, blk128, 0, stream>>>(agg, stats);
        stats_final_kernel  <<<1, blk128, 0, stream>>>(stats);
        bn_apply_kernel<<<NN * HID / 256, blk256, 0, stream>>>(
            agg, stats, G[layer], BE[layer], add, hout, hdup);
    };

    conv_bn(X, DIN, 0, nullptr, h, resid);                 // layer 1 (+keep residual)
    conv_bn(h, HID, 1, nullptr, h, nullptr);               // layer 2
    gemm_wmma_kernel<0, float><<<dim3(HID / 16, NN / 64), blk128, 0, stream>>>(
        resid, resW, resB, rlin, NN, HID, HID);            // residual projection
    conv_bn(h, HID, 2, rlin, h, nullptr);                  // layer 3 (+residual)
    conv_bn(h, HID, 3, nullptr, h, nullptr);               // layer 4

    // qkv = h @ in_proj_w^T + in_proj_b   [NN x 384], stored f16 for attention
    gemm_wmma_kernel<1, _Float16><<<dim3(384 / 16, NN / 64), blk128, 0, stream>>>(
        h, inW, inB, qh, NN, 384, HID);

    // flash attention -> ctx [NN x 128]
    attn_kernel<<<dim3(NN / 128, NHEAD), blk128, 0, stream>>>(qh, ctx);

    // out_proj: agg = ctx @ out_proj_w^T + out_proj_b
    gemm_wmma_kernel<1, float><<<dim3(HID / 16, NN / 64), blk128, 0, stream>>>(
        ctx, opW, opB, agg, NN, HID, HID);

    // final: OUT = agg @ out_W + out_b    [NN x 16]
    gemm_wmma_kernel<0, float><<<dim3(DOUT / 16, NN / 64), blk128, 0, stream>>>(
        agg, outW, outB, OUT, NN, DOUT, HID);
}
